// SparseBlock_Conv2d_BN_ReLU_14663018348980
// MI455X (gfx1250) — matrix operand order repaired and checked by reading, not compiled, not run on hardware
//
#include <hip/hip_runtime.h>
#include <hip/hip_bf16.h>

// ---- problem constants ----
#define HH   768
#define WW   768
#define CIN  64
#define COUT 64
#define NBLK 256
#define BOUT 48
#define BSTRD 48

// ---- WMMA vector types (CDNA5 gfx1250, wave32) ----
typedef __attribute__((ext_vector_type(4)))  float        v4f;
typedef __attribute__((ext_vector_type(8)))  float        v8f;
typedef __attribute__((ext_vector_type(8)))  __bf16       v8bf;
typedef __attribute__((ext_vector_type(16))) __bf16       v16bf;
typedef __attribute__((ext_vector_type(4)))  unsigned int v4u;
typedef __attribute__((ext_vector_type(8)))  unsigned int v8u;

// weight tile in ws/LDS: Wr[tap(9)][g(4)][cout(64)][pad 24 (16 used)] bf16
#define WPAD      24
#define W_ELEMS   (9*4*COUT*WPAD)        // 55296 bf16
#define W_BYTES   (W_ELEMS*2)            // 110592 B (divisible by 8)
// patch strip in LDS: [18 rows][50 cols][pitch 72 (64 used)] bf16
#define P_ROWS    18
#define P_COLS    50
#define P_PITCH   72
#define P_BYTES   (P_ROWS*P_COLS*P_PITCH*2)  // 129600 B
#define LDS_BYTES (W_BYTES + P_BYTES)        // 240192 B <= 320KB WGP LDS

// --------------------------------------------------------------------------
// Kernel 1: reorder conv weights -> bf16 WMMA-B layout; fold bias+BN.
// Wr[((tap*4+g)*64+cout)*24 + j] = conv_w[cout][g*16+j][ky][kx]
// scale[c] = gamma/sqrt(var+eps); shift[c] = beta + (bias - mean)*scale
// --------------------------------------------------------------------------
__global__ void prep_kernel(const float* __restrict__ conv_w,
                            const float* __restrict__ conv_b,
                            const float* __restrict__ gamma,
                            const float* __restrict__ beta,
                            const float* __restrict__ rmean,
                            const float* __restrict__ rvar,
                            __bf16* __restrict__ wr,
                            float* __restrict__ scale,
                            float* __restrict__ shift) {
    int idx = blockIdx.x * 256 + threadIdx.x;
    if (idx < 9*4*64*16) {                 // 36864 weight elements
        int j    = idx & 15;
        int cout = (idx >> 4) & 63;
        int g    = (idx >> 10) & 3;
        int tap  = idx >> 12;              // 0..8
        int ky = tap / 3, kx = tap % 3;
        int cin = g * 16 + j;
        float v = conv_w[((cout*CIN + cin)*3 + ky)*3 + kx];
        wr[((tap*4 + g)*COUT + cout)*WPAD + j] = (__bf16)v;
    } else if (idx < 9*4*64*16 + COUT) {
        int c = idx - 9*4*64*16;
        float inv = gamma[c] / sqrtf(rvar[c] + 1e-3f);
        scale[c] = inv;
        shift[c] = beta[c] + (conv_b[c] - rmean[c]) * inv;
    }
}

// --------------------------------------------------------------------------
// Kernel 2: zero-fill the 48x48x64 regions of all INACTIVE tiles.
// Non-temporal stores: don't pollute L2 with 151MB of zeros.
// --------------------------------------------------------------------------
__global__ __launch_bounds__(256) void zero_inactive_kernel(
        const int* __restrict__ abi, float* __restrict__ y) {
    const int b  = blockIdx.x;           // 0..511
    const int n  = b >> 8;
    const int bh = (b >> 4) & 15;
    const int bw = b & 15;
    __shared__ int active;
    if (threadIdx.x == 0) active = 0;
    __syncthreads();
    const int t = threadIdx.x;
    if (t < NBLK) {
        if (abi[t*3] == n && abi[t*3+1] == bh && abi[t*3+2] == bw) active = 1;
    }
    __syncthreads();
    if (active) return;                  // conv kernel writes this region
    const v4f z = {0.f, 0.f, 0.f, 0.f};
    for (int rc = t; rc < COUT*BOUT; rc += 256) {
        const int c = rc / BOUT, r = rc % BOUT;
        v4f* p = (v4f*)(y + (((size_t)(n*COUT + c)*HH + bh*BSTRD + r)*WW
                             + bw*BSTRD));
#pragma unroll
        for (int i = 0; i < 12; ++i) __builtin_nontemporal_store(z, &p[i]);
    }
}

// --------------------------------------------------------------------------
// Kernel 3: WMMA implicit-GEMM conv + BN + ReLU for active blocks.
// grid.x = 256 blocks * 3 row-strips. 256 threads = 8 wave32s.
// Weights arrive via Tensor Data Mover (TDM) DMA; patch via vectorized
// fp32->bf16 staging. GEMM: M=16 pixels, N=4x16 couts, K=576.
// --------------------------------------------------------------------------
__global__ __launch_bounds__(256, 1) void conv_wmma_kernel(
        const float*  __restrict__ x,
        const int*    __restrict__ abi,
        const __bf16* __restrict__ wr,
        const float*  __restrict__ scale,
        const float*  __restrict__ shift,
        float*        __restrict__ y) {
    extern __shared__ __align__(16) char smem[];
    __bf16* wlds = (__bf16*)smem;                 // weights, 110592 B
    __bf16* plds = (__bf16*)(smem + W_BYTES);     // input patch strip

    const int bid   = blockIdx.x / 3;
    const int strip = blockIdx.x % 3;
    const int n  = abi[bid*3 + 0];
    const int bh = abi[bid*3 + 1];
    const int bw = abi[bid*3 + 2];
    const int tid = threadIdx.x;
    const int wv   = tid >> 5;
    const int lane = tid & 31;

    // ---- wave 0: TDM DMA of the pre-packed bf16 weight tile ws -> LDS ----
    // D# per CDNA5 ISA: 1-row tile of 13824 x 8-byte
    // elements. Other waves overlap patch staging with the DMA.
    if (wv == 0) {
        const unsigned lds_base = (unsigned)(uintptr_t)(void*)wlds; // LDS offset
        const unsigned long long ga = (unsigned long long)(uintptr_t)wr;
        const unsigned TD0 = W_BYTES / 8;            // 13824 elements of 8B
        v4u g0;
        g0[0] = 1u;                                   // count=1 (valid user D#)
        g0[1] = lds_base;                             // lds_addr
        g0[2] = (unsigned)(ga & 0xffffffffu);         // global_addr[31:0]
        g0[3] = (unsigned)((ga >> 32) & 0x1ffffffu)   // global_addr[56:32]
                | (2u << 30);                         // type=2 ("image")
        v8u g1;
        g1[0] = 3u << 16;                             // data_size=3 (8 bytes)
        g1[1] = (TD0 & 0xffffu) << 16;                // tensor_dim0[15:0]
        g1[2] = (TD0 >> 16) | (1u << 16);             // tensor_dim0[31:16] | tensor_dim1=1
        g1[3] = (TD0 & 0xffffu) << 16;                // tile_dim0 = 13824
        g1[4] = 1u;                                   // tile_dim1 = 1
        g1[5] = TD0;                                  // tensor_dim0_stride lo
        g1[6] = 0u;
        g1[7] = 0u;
        asm volatile("tensor_load_to_lds %0, %1" :: "s"(g0), "s"(g1) : "memory");
        __builtin_amdgcn_s_wait_tensorcnt(0);
    }

    // ---- stage input halo strip (fp32 -> bf16), rows oh-1..oh+16, cols ow-1..ow+48
    {
        const int h0 = bh*BSTRD + strip*16 - 1;
        const int w0 = bw*BSTRD - 1;
        for (int p = tid; p < P_ROWS*CIN; p += 256) {
            const int rr = p >> 6;        // 0..17
            const int c  = p & 63;        // cin
            const int gh = h0 + rr;
            __bf16* drow = plds + (rr*P_COLS)*P_PITCH + c;   // +cc*P_PITCH per col
            if (gh < 0 || gh >= HH) {
                for (int cc = 0; cc < P_COLS; ++cc)
                    drow[cc*P_PITCH] = (__bf16)0.f;
            } else {
                const float* src = x + ((size_t)(n*CIN + c)*HH + gh)*WW;
                float v0 = (w0 >= 0) ? src[w0] : 0.f;        // left halo col
                drow[0] = (__bf16)v0;
                const v4f* s4 = (const v4f*)(src + w0 + 1);  // 16B aligned
#pragma unroll
                for (int i = 0; i < 12; ++i) {                // cols 1..48
                    v4f q = s4[i];
                    drow[(1 + 4*i + 0)*P_PITCH] = (__bf16)q.x;
                    drow[(1 + 4*i + 1)*P_PITCH] = (__bf16)q.y;
                    drow[(1 + 4*i + 2)*P_PITCH] = (__bf16)q.z;
                    drow[(1 + 4*i + 3)*P_PITCH] = (__bf16)q.w;
                }
                const int gw49 = w0 + 49;                     // right halo col
                float v49 = (gw49 < WW) ? src[gw49] : 0.f;
                drow[49*P_PITCH] = (__bf16)v49;
            }
        }
    }
    __syncthreads();

    const int half = lane >> 4;   // 0: lanes 0-15, 1: lanes 16-31
    const int lm   = lane & 15;

    // 48 M-tiles (16 rows * 3 col-segments); each wave does 6, computing all
    // 4 cout tiles at once so every A fetch feeds 4 independent WMMAs.
    for (int mt = wv; mt < 48; mt += 8) {
        const int orow = mt / 3;          // 0..15 within strip
        const int cseg = mt % 3;          // 16-col segment

        v8f acc[4] = {};
#pragma unroll
        for (int ky = 0; ky < 3; ++ky) {
            const int rr = orow + ky;
#pragma unroll
            for (int kx = 0; kx < 3; ++kx) {
                const int tap = ky*3 + kx;
                // A row M=lm -> pixel (orow, cseg*16+lm); tap shifts by (ky,kx)
                const __bf16* apix = plds + (rr*P_COLS + (cseg*16 + kx + lm))*P_PITCH;
#pragma unroll
                for (int g2 = 0; g2 < 2; ++g2) {     // K chunks of 32 cin
                    // A 16x32, contiguous k order (same as B): half0 holds K 0-15,
                    //                        half1 holds K 16-31
                    const int aoff = g2*32 + half*16;
                    v8bf alo = *(const v8bf*)(apix + aoff);
                    v8bf ahi = *(const v8bf*)(apix + aoff + 8);
                    v16bf A;
#pragma unroll
                    for (int i2 = 0; i2 < 8; ++i2) { A[i2] = alo[i2]; A[i2+8] = ahi[i2]; }
                    // B 32x16: half0 rows K0-15 (group 2*g2), half1 K16-31
                    const int g = g2*2 + half;
                    const __bf16* brow = wlds + ((tap*4 + g)*COUT + lm)*WPAD;
#pragma unroll
                    for (int nt = 0; nt < 4; ++nt) {
                        const __bf16* bptr = brow + (nt*16)*WPAD;
                        v8bf blo = *(const v8bf*)(bptr);
                        v8bf bhi = *(const v8bf*)(bptr + 8);
                        v16bf B;
#pragma unroll
                        for (int i2 = 0; i2 < 8; ++i2) { B[i2] = blo[i2]; B[i2+8] = bhi[i2]; }
                        acc[nt] = __builtin_amdgcn_wmma_f32_16x16x32_bf16(
                                false, A, false, B, (short)0, acc[nt], false, false);
                    }
                }
            }
        }

        // fused bias+BN+ReLU, then store: acc[nt][i] is row M=i+8*half (=col
        // offset), col N=lm (=cout nt*16+lm)
        const int oh = bh*BSTRD + strip*16 + orow;
        const int ow = bw*BSTRD + cseg*16 + half*8;   // M index maps to col
#pragma unroll
        for (int nt = 0; nt < 4; ++nt) {
            const int cout = nt*16 + lm;
            const float sA = scale[cout];
            const float sB = shift[cout];
            v8f a = acc[nt];
#pragma unroll
            for (int i2 = 0; i2 < 8; ++i2) {
                float v = a[i2]*sA + sB;
                a[i2] = v > 0.f ? v : 0.f;
            }
            float* dst = y + (((size_t)(n*COUT + cout)*HH + oh)*WW + ow);
            v4f lo = {a[0], a[1], a[2], a[3]};
            v4f hi = {a[4], a[5], a[6], a[7]};
            *(v4f*)(dst)     = lo;
            *(v4f*)(dst + 4) = hi;
        }
    }
}

// --------------------------------------------------------------------------
extern "C" void kernel_launch(void* const* d_in, const int* in_sizes, int n_in,
                              void* d_out, int out_size, void* d_ws, size_t ws_size,
                              hipStream_t stream) {
    const float* x      = (const float*)d_in[0];
    const float* conv_w = (const float*)d_in[1];
    const float* conv_b = (const float*)d_in[2];
    const float* gamma  = (const float*)d_in[3];
    const float* beta   = (const float*)d_in[4];
    const float* rmean  = (const float*)d_in[5];
    const float* rvar   = (const float*)d_in[6];
    const int*   abi    = (const int*)d_in[7];
    float* y = (float*)d_out;

    __bf16* wr   = (__bf16*)d_ws;                       // 110592 B
    float*  scl  = (float*)((char*)d_ws + W_BYTES);     // 64 floats
    float*  shf  = scl + COUT;                          // 64 floats

    prep_kernel<<<(9*4*64*16 + COUT + 255)/256, 256, 0, stream>>>(
        conv_w, conv_b, gamma, beta, rmean, rvar, wr, scl, shf);

    zero_inactive_kernel<<<2*16*16, 256, 0, stream>>>(abi, y);

    // CDNA5 allows up to 320KB LDS per workgroup; request 240KB dynamic LDS.
    (void)hipFuncSetAttribute(reinterpret_cast<const void*>(conv_wmma_kernel),
                              hipFuncAttributeMaxDynamicSharedMemorySize,
                              LDS_BYTES);
    conv_wmma_kernel<<<NBLK*3, 256, LDS_BYTES, stream>>>(x, abi, wr, scl, shf, y);
}
